// ASM_FineEnhancement_71691594105134
// MI455X (gfx1250) — compile-verified, matrix operand order checked by reading, not run on hardware
//
#include <hip/hip_runtime.h>
#include <hip/hip_bf16.h>

typedef __attribute__((ext_vector_type(16))) _Float16 v16h;
typedef __attribute__((ext_vector_type(8)))  float    v8f;

#define B_    4
#define CH_   256
#define H_    256
#define W_    256
#define CP_   16
#define FP_   8
#define LC_   256      // (H/CP)*(W/CP) coarse patches per batch
#define KTOP_ 64       // int(0.25 * 256)

// ---------------------------------------------------------------------------
// Kernel 1: repack conv weights fp32 -> packed f16 pairs, layout
//   WB[t][kb][co][kp]  (t=tap 0..8, kb=ci-block-of-32 0..7, co 0..255, kp 0..15)
//   packing ci = kb*32 + 2*kp (+1).  A lane's B fragment (K-pairs
//   {0..3}+8k / {4..7}+8k) is then two contiguous uint4 (b128) loads.
// ---------------------------------------------------------------------------
__global__ void k_prep_w(const float* __restrict__ w, unsigned* __restrict__ wb) {
    int idx = blockIdx.x * blockDim.x + threadIdx.x;   // 9*8*256*16 = 294912
    int kp = idx & 15;
    int co = (idx >> 4) & 255;
    int kb = (idx >> 12) & 7;
    int tp = idx >> 15;
    int dy = tp / 3, dx = tp % 3;
    int ci0 = kb * 32 + 2 * kp;
    float w0 = w[(((size_t)co * CH_ + ci0    ) * 3 + dy) * 3 + dx];
    float w1 = w[(((size_t)co * CH_ + ci0 + 1) * 3 + dy) * 3 + dx];
    union { _Float16 h[2]; unsigned u; } pk;
    pk.h[0] = (_Float16)w0; pk.h[1] = (_Float16)w1;
    wb[idx] = pk.u;
}

// ---------------------------------------------------------------------------
// Kernel 2: copy x -> out (identity for non-selected regions) and compute
// per-coarse-patch mean(|x|) in a single pass over the 256MB image.
// One block per (batch, coarse patch).
// ---------------------------------------------------------------------------
__global__ void k_score_copy(const float* __restrict__ x, float* __restrict__ out,
                             float* __restrict__ scores) {
    int bid = blockIdx.x;              // B_*LC_
    int b  = bid >> 8;
    int lc = bid & 255;
    int gy0 = (lc >> 4) * CP_;
    int gx0 = (lc & 15) * CP_;
    size_t imgBase = (size_t)b * CH_ * H_ * W_;
    float acc = 0.f;
    for (int idx = threadIdx.x; idx < CH_ * CP_ * CP_; idx += 256) {
        int ci = idx >> 8;
        int p  = idx & 255;
        int y = p >> 4, xx = p & 15;
        size_t g = imgBase + ((size_t)ci * H_ + gy0 + y) * W_ + gx0 + xx;
        float v = x[g];
        out[g] = v;
        acc += fabsf(v);
    }
    __shared__ float red[256];
    red[threadIdx.x] = acc;
    __syncthreads();
    for (int s = 128; s > 0; s >>= 1) {
        if (threadIdx.x < s) red[threadIdx.x] += red[threadIdx.x + s];
        __syncthreads();
    }
    if (threadIdx.x == 0)
        scores[b * LC_ + lc] = red[0] * (1.0f / (CH_ * CP_ * CP_));
}

// ---------------------------------------------------------------------------
// Kernel 3: top-64 per batch via exact ranking (256 scores, ties -> lower idx
// first, matching jax top_k). Ranks are unique => deterministic compaction.
// ---------------------------------------------------------------------------
__global__ void k_topk(const float* __restrict__ scores, int* __restrict__ sel) {
    int b = blockIdx.x;
    int t = threadIdx.x;
    __shared__ float s[LC_];
    s[t] = scores[b * LC_ + t];
    __syncthreads();
    float mine = s[t];
    int rank = 0;
    for (int j = 0; j < LC_; ++j) {
        float o = s[j];
        rank += (o > mine) || (o == mine && j < t);
    }
    if (rank < KTOP_) sel[b * KTOP_ + rank] = t;
}

// ---------------------------------------------------------------------------
// Kernel 4: per-fine-patch 3x3 conv (256->256 ch) + bias + ReLU via
// v_wmma_f32_16x16x32_f16.  One 256-thread WG (8 wave32) per fine patch.
//   LDS: zero-padded 10x10 patch, f16, layout [spatial s][ci-pair cp] so A
//   fragments are two ds_load_b128 per lane.
//   GEMM: M=64 positions (4 M-tiles) x N=256 out-ch (2 N-tiles per wave),
//   K = 9 taps * 8 ci-blocks = 72 WMMA k-steps, fp32 accumulators.
// ---------------------------------------------------------------------------
__global__ void __launch_bounds__(256)
k_conv(const float* __restrict__ x, const unsigned* __restrict__ wb,
       const float* __restrict__ bias, const int* __restrict__ sel,
       float* __restrict__ out) {
    __shared__ unsigned lds[100 * 128];     // 10x10 spatial x 128 ci-pairs = 50KB

    int bid = blockIdx.x;                   // B_*KTOP_*4 = 1024
    int b  = bid >> 8;
    int j  = bid & 255;
    int si = j >> 2;
    int q  = j & 3;                         // quadrant of coarse patch
    int lc = sel[b * KTOP_ + si];
    int rc = lc >> 4, cc = lc & 15;
    int gy0 = (2 * rc + (q >> 1)) * FP_;
    int gx0 = (2 * cc + (q & 1)) * FP_;
    size_t imgBase = (size_t)b * CH_ * H_ * W_;
    int tid = threadIdx.x;

    // zero the padded LDS image (border stays zero)
    for (int i = tid; i < 100 * 128; i += 256) lds[i] = 0u;
    __syncthreads();

    // stage 8x8x256ch patch, fp32 -> packed f16 pairs
    #pragma unroll 4
    for (int it = 0; it < 32; ++it) {
        int idx = it * 256 + tid;           // 128 cp * 64 positions
        int pi = idx & 63;                  // spatial fastest -> coalesced rows
        int cp = idx >> 6;
        int y = pi >> 3, xx = pi & 7;
        size_t g = imgBase + ((size_t)(2 * cp) * H_ + gy0 + y) * W_ + gx0 + xx;
        float f0 = x[g];
        float f1 = x[g + (size_t)H_ * W_];
        union { _Float16 h[2]; unsigned u; } pk;
        pk.h[0] = (_Float16)f0; pk.h[1] = (_Float16)f1;
        int s = (y + 1) * 10 + (xx + 1);
        lds[s * 128 + cp] = pk.u;
    }
    __syncthreads();

    int wave = tid >> 5;
    int lane = tid & 31;
    int l16  = lane & 15;
    int lh   = lane >> 4;                   // K-half select per ISA A/B layout

    v8f zero = {};
    v8f acc[2][4];
    #pragma unroll
    for (int a = 0; a < 2; ++a)
        #pragma unroll
        for (int m = 0; m < 4; ++m) acc[a][m] = zero;

    const uint4* lds4 = (const uint4*)lds;
    const uint4* wb4  = (const uint4*)wb;

    int py[4], px[4];
    #pragma unroll
    for (int mt = 0; mt < 4; ++mt) {        // A-row p = mt*16 + l16
        int p = mt * 16 + l16;
        py[mt] = p >> 3; px[mt] = p & 7;
    }
    int co0 = wave * 32 + l16;              // wave covers out-ch tiles 2w, 2w+1

    #pragma unroll 1
    for (int tp = 0; tp < 9; ++tp) {
        int dy = tp / 3, dx = tp % 3;
        for (int kb = 0; kb < 8; ++kb) {
            union AF { uint4 q[2]; v16h h; } a[4];
            #pragma unroll
            for (int mt = 0; mt < 4; ++mt) {
                int s = (py[mt] + dy) * 10 + (px[mt] + dx);
                int base = s * 32 + kb * 4 + lh;     // uint4 index
                a[mt].q[0] = lds4[base];             // K-pairs {0..3}+4*lh
                a[mt].q[1] = lds4[base + 2];         // K-pairs {8..11}+4*lh
            }
            union BF { uint4 q[2]; v16h h; } bm[2];
            #pragma unroll
            for (int nt = 0; nt < 2; ++nt) {
                int co = co0 + nt * 16;
                int base = ((tp * 8 + kb) * 256 + co) * 4 + lh;
                bm[nt].q[0] = wb4[base];
                bm[nt].q[1] = wb4[base + 2];
            }
            #pragma unroll
            for (int nt = 0; nt < 2; ++nt)
                #pragma unroll
                for (int mt = 0; mt < 4; ++mt)
                    acc[nt][mt] = __builtin_amdgcn_wmma_f32_16x16x32_f16(
                        false, a[mt].h, false, bm[nt].h,
                        (short)0, acc[nt][mt], false, false);
        }
    }

    // epilogue: bias + ReLU, scatter into output image.
    // D layout: lane(0-15)=col l16, VGPR r = row r (lanes<16) / r+8 (lanes>=16)
    #pragma unroll
    for (int nt = 0; nt < 2; ++nt) {
        int co = co0 + nt * 16;
        float bv = bias[co];
        size_t chanBase = imgBase + (size_t)co * H_ * W_;
        #pragma unroll
        for (int mt = 0; mt < 4; ++mt) {
            #pragma unroll
            for (int r = 0; r < 8; ++r) {
                int p = mt * 16 + r + lh * 8;
                int y = p >> 3, xx = p & 7;
                float v = acc[nt][mt][r] + bv;
                v = v > 0.f ? v : 0.f;
                out[chanBase + (size_t)(gy0 + y) * W_ + gx0 + xx] = v;
            }
        }
    }
}

// ---------------------------------------------------------------------------
extern "C" void kernel_launch(void* const* d_in, const int* in_sizes, int n_in,
                              void* d_out, int out_size, void* d_ws, size_t ws_size,
                              hipStream_t stream) {
    const float* x      = (const float*)d_in[0];
    const float* conv_w = (const float*)d_in[1];
    const float* conv_b = (const float*)d_in[2];
    float* out = (float*)d_out;

    char* ws = (char*)d_ws;
    float*    scores = (float*)(ws + 0);        // 4*256 f32
    int*      sel    = (int*)(ws + 4096);       // 4*64 i32
    unsigned* wbuf   = (unsigned*)(ws + 8192);  // 9*8*256*16 u32 = 1.18MB

    k_prep_w<<<(9 * 8 * 256 * 16) / 256, 256, 0, stream>>>(conv_w, wbuf);
    k_score_copy<<<B_ * LC_, 256, 0, stream>>>(x, out, scores);
    k_topk<<<B_, 256, 0, stream>>>(scores, sel);
    k_conv<<<B_ * KTOP_ * 4, 256, 0, stream>>>(x, wbuf, conv_b, sel, out);
}
